// TFRobertaSelfAttention_33371895890168
// MI455X (gfx1250) — compile-verified, matrix operand order checked by reading, not run on hardware
//
#include <hip/hip_runtime.h>

typedef __attribute__((ext_vector_type(16))) _Float16 v16h;
typedef __attribute__((ext_vector_type(8)))  float    v8f;
typedef __attribute__((ext_vector_type(8)))  _Float16 h8;
typedef __attribute__((ext_vector_type(4)))  int      v4i;

#define S_LEN 2048
#define D_DIM 1024
#define NH    16
#define HD    64

// ---- CDNA5 async global->LDS copy (probe-verified: 4 args, v4i* params) ----
#if defined(__gfx1250__) && __has_builtin(__builtin_amdgcn_global_load_async_to_lds_b128)
#define ASYNC_COPY 1
#define ASYNC_LDS_B128(gsrc, ldst) \
    __builtin_amdgcn_global_load_async_to_lds_b128((v4i*)(gsrc), (v4i*)(ldst), 0, 0)
#if __has_builtin(__builtin_amdgcn_s_wait_asynccnt)
#define WAIT_ASYNC() __builtin_amdgcn_s_wait_asynccnt(0)
#else
#define WAIT_ASYNC() asm volatile("s_wait_asynccnt 0" ::: "memory")
#endif
#else
#define ASYNC_COPY 0
#endif

// ---------------------------------------------------------------------------
// Kernel 1: fused QKV projection.
//   C[M,N] = X[M,K] @ W[K,N] + bias  (M = B*S = 4096, K = N = 1024)
// fp32 inputs converted to f16 while filling LDS; f32 WMMA accumulation.
// W tile stored COLUMN-MAJOR in LDS so B fragments are contiguous b128 reads.
// Q,K outputs -> [B,H,S,HD]; V output -> [B,H,HD,S] (transposed per head,
// expressed purely via strides so the epilogue stays branch-free).
// ---------------------------------------------------------------------------
__global__ __launch_bounds__(256) void qkv_gemm_kernel(
    const float* __restrict__ X,
    const float* __restrict__ W0, const float* __restrict__ b0,
    const float* __restrict__ W1, const float* __restrict__ b1,
    const float* __restrict__ W2, const float* __restrict__ b2,
    _Float16* __restrict__ Qo, _Float16* __restrict__ Ko, _Float16* __restrict__ Vo)
{
    const int z = blockIdx.z;
    const float* W    = (z == 0) ? W0 : (z == 1) ? W1 : W2;
    const float* bias = (z == 0) ? b0 : (z == 1) ? b1 : b2;
    _Float16* Out     = (z == 0) ? Qo : (z == 1) ? Ko : Vo;
    const size_t sStr = (z == 2) ? 1 : HD;       // seq-index stride in output
    const size_t dStr = (z == 2) ? S_LEN : 1;    // headdim-index stride

    __shared__ _Float16 sA[128][40];    // A tile, row-major [m][k], 80B stride
    __shared__ _Float16 sBT[128][40];   // B tile, col-major [n][k], 80B stride

    const int tid  = threadIdx.x;
    const int lane = tid & 31;
    const int wave = tid >> 5;
    const int wm   = wave >> 2;          // 0..1
    const int wn   = wave & 3;           // 0..3
    const int bm   = blockIdx.y * 128;
    const int bn   = blockIdx.x * 128;
    const int l16  = lane & 15;
    const int hsel = (lane < 16) ? 0 : 8;   // K-half select for 16-bit fragments
    const int rhi  = (lane >> 4) << 3;      // C-layout row offset (0 or 8)

    v8f acc[4][2];
    #pragma unroll
    for (int t = 0; t < 4; ++t)
        #pragma unroll
        for (int u = 0; u < 2; ++u) {
            float bv = bias[bn + wn * 32 + u * 16 + l16];
            #pragma unroll
            for (int i = 0; i < 8; ++i) acc[t][u][i] = bv;
        }

    for (int k0 = 0; k0 < D_DIM; k0 += 32) {
        // unconditional wrap-around prefetch: no loop peeling
        {
            int kn = (k0 + 32) & (D_DIM - 1);
            __builtin_prefetch(X + (size_t)(bm + (tid >> 1)) * D_DIM + kn + (tid & 1) * 16, 0, 1);
            __builtin_prefetch(W + (size_t)(kn + (tid >> 3)) * D_DIM + bn + (tid & 7) * 16, 0, 1);
        }
        // ---- fill LDS (fp32 global -> f16 LDS) ----
        #pragma unroll
        for (int it = 0; it < 4; ++it) {              // A: 128x32 = 1024 float4
            int idx = it * 256 + tid;
            int row = idx >> 3;
            int c4  = (idx & 7) << 2;
            float4 v = *(const float4*)(X + (size_t)(bm + row) * D_DIM + k0 + c4);
            sA[row][c4 + 0] = (_Float16)v.x;
            sA[row][c4 + 1] = (_Float16)v.y;
            sA[row][c4 + 2] = (_Float16)v.z;
            sA[row][c4 + 3] = (_Float16)v.w;
        }
        #pragma unroll
        for (int it = 0; it < 4; ++it) {              // B: 32x128 = 1024 float4
            int idx = it * 256 + tid;
            int row = idx >> 5;                       // k within tile
            int c4  = (idx & 31) << 2;                // n within tile
            float4 v = *(const float4*)(W + (size_t)(k0 + row) * D_DIM + bn + c4);
            sBT[c4 + 0][row] = (_Float16)v.x;         // transposed store
            sBT[c4 + 1][row] = (_Float16)v.y;
            sBT[c4 + 2][row] = (_Float16)v.z;
            sBT[c4 + 3][row] = (_Float16)v.w;
        }
        __syncthreads();

        // ---- fragments (all contiguous -> ds_load_b128) + WMMA ----
        v16h bfr[2];
        #pragma unroll
        for (int u = 0; u < 2; ++u) {
            int col = wn * 32 + u * 16 + l16;
            #pragma unroll
            for (int i = 0; i < 8; ++i) {
                bfr[u][i]     = sBT[col][hsel + i];
                bfr[u][8 + i] = sBT[col][16 + hsel + i];
            }
        }
        #pragma unroll
        for (int t = 0; t < 4; ++t) {
            v16h a;
            int row = wm * 64 + t * 16 + l16;
            #pragma unroll
            for (int i = 0; i < 8; ++i) {
                a[i]     = sA[row][hsel + i];
                a[8 + i] = sA[row][16 + hsel + i];
            }
            acc[t][0] = __builtin_amdgcn_wmma_f32_16x16x32_f16(
                false, a, false, bfr[0], (short)0, acc[t][0], false, false);
            acc[t][1] = __builtin_amdgcn_wmma_f32_16x16x32_f16(
                false, a, false, bfr[1], (short)0, acc[t][1], false, false);
        }
        __syncthreads();
    }

    // ---- epilogue: branch-free strided scatter to f16 workspace ----
    // 128-row block never straddles a batch (2048 % 128 == 0)
    const int bb    = bm >> 11;
    const int sBase = (bm & (S_LEN - 1)) + wm * 64 + rhi;
    #pragma unroll
    for (int t = 0; t < 4; ++t)
        #pragma unroll
        for (int u = 0; u < 2; ++u) {
            int colg = bn + wn * 32 + u * 16 + l16;
            int h  = colg >> 6;        // / HD
            int hd = colg & 63;
            _Float16* outHead = Out + (size_t)(bb * NH + h) * (S_LEN * HD) + (size_t)hd * dStr;
            #pragma unroll
            for (int i = 0; i < 8; ++i) {
                int s = sBase + t * 16 + i;
                outHead[(size_t)s * sStr] = (_Float16)acc[t][u][i];
            }
        }
}

// ---------------------------------------------------------------------------
// Kernel 2: flash attention.
// Block = 256 threads (8 waves) owns 128 query rows of one (b,h);
// each wave owns a 16-row strip. Per 64-key tile:
//   scores(16x64) = Q @ K^T (8 WMMA), online softmax in registers,
//   P re-laid-out via per-wave LDS, ctx += P @ V (8 WMMA).
// K tile [key][d] and V^T tile [d][key] are straight contiguous copies ->
// async global->LDS b128 (ASYNCcnt) when available, b128 copies otherwise.
// ---------------------------------------------------------------------------
__global__ __launch_bounds__(256) void attn_kernel(
    const _Float16* __restrict__ Q,    // [B,H,S,HD]
    const _Float16* __restrict__ Kg,   // [B,H,S,HD]
    const _Float16* __restrict__ Vg,   // [B,H,HD,S]  (pre-transposed)
    const float* __restrict__ mask,    // [B,1,1,S] -> flat [B,S]
    float* __restrict__ Out)           // [B,S,D] fp32
{
    __shared__ _Float16 sQ[128][72];      // [q][d]   144B stride
    __shared__ _Float16 sK[64][72];       // [key][d]
    __shared__ _Float16 sVT[HD][72];      // [d][key]
    __shared__ _Float16 sP[8][16][72];    // per-wave P staging [q][key]

    const int tid  = threadIdx.x;
    const int lane = tid & 31;
    const int wave = tid >> 5;
    const int l16  = lane & 15;
    const int hsel = (lane < 16) ? 0 : 8;
    const int rhi  = (lane >> 4) << 3;

    const int q0 = blockIdx.x * 128;
    const int bh = blockIdx.y;
    const int b  = bh >> 4;
    const int h  = bh & 15;

    const _Float16* Qp  = Q  + ((size_t)bh * S_LEN + q0) * HD;
    const _Float16* Kp  = Kg + (size_t)bh * S_LEN * HD;
    const _Float16* VpT = Vg + (size_t)bh * HD * S_LEN;
    const float*    mp  = mask + (size_t)b * S_LEN;

    // load Q tile as 8-half chunks, pre-scaled by 1/sqrt(64) (v_pk_mul_f16)
    #pragma unroll
    for (int it = 0; it < 4; ++it) {
        int cc = it * 256 + tid;              // chunk index, rows contiguous
        int r = cc >> 3, c8 = (cc & 7) * 8;
        h8 v = *(const h8*)(Qp + (size_t)cc * 8);
        v = v * (_Float16)0.125f;
        *(h8*)&sQ[r][c8] = v;
    }

    v8f o[4];
    #pragma unroll
    for (int c = 0; c < 4; ++c)
        #pragma unroll
        for (int i = 0; i < 8; ++i) o[c][i] = 0.0f;

    float mr[8], lr[8];
    #pragma unroll
    for (int i = 0; i < 8; ++i) { mr[i] = -3.0e30f; lr[i] = 0.0f; }

    for (int k0 = 0; k0 < S_LEN; k0 += 64) {
        __syncthreads();                       // protect previous-tile readers
        // ---- K/V tile fill: 64x64 halves = 512 x 16B chunks each ----
        #pragma unroll
        for (int it = 0; it < 2; ++it) {
            int cc = it * 256 + tid;
            int r = cc >> 3, c8 = (cc & 7) * 8;
#if ASYNC_COPY
            ASYNC_LDS_B128(Kp + (size_t)(k0 + r) * HD + c8, &sK[r][c8]);
            ASYNC_LDS_B128(VpT + (size_t)r * S_LEN + k0 + c8, &sVT[r][c8]);
#else
            *(h8*)&sK[r][c8]  = *(const h8*)(Kp + (size_t)(k0 + r) * HD + c8);
            *(h8*)&sVT[r][c8] = *(const h8*)(VpT + (size_t)r * S_LEN + k0 + c8);
#endif
        }
#if ASYNC_COPY
        WAIT_ASYNC();
#endif
        // unconditional wrap-around prefetch: no loop peeling
        {
            int kn = (k0 + 64) & (S_LEN - 1);
            __builtin_prefetch(Kp + (size_t)(kn + (tid >> 2)) * HD + (tid & 3) * 16, 0, 1);
            __builtin_prefetch(VpT + (size_t)(tid >> 2) * S_LEN + kn + (tid & 3) * 16, 0, 1);
        }
        __syncthreads();

        // ---- scores = Q(16x64) @ K^T(64x64) ----
        v8f sc[4];
        #pragma unroll
        for (int c = 0; c < 4; ++c)
            #pragma unroll
            for (int i = 0; i < 8; ++i) sc[c][i] = 0.0f;

        #pragma unroll
        for (int ks = 0; ks < 2; ++ks) {
            v16h a;
            int row = wave * 16 + l16;
            int kb  = ks * 32;
            #pragma unroll
            for (int i = 0; i < 8; ++i) {
                a[i]     = sQ[row][kb + hsel + i];
                a[8 + i] = sQ[row][kb + 16 + hsel + i];
            }
            #pragma unroll
            for (int c = 0; c < 4; ++c) {
                v16h bf;
                int col = c * 16 + l16;            // key column
                #pragma unroll
                for (int i = 0; i < 8; ++i) {
                    bf[i]     = sK[col][kb + hsel + i];        // contiguous
                    bf[8 + i] = sK[col][kb + 16 + hsel + i];
                }
                sc[c] = __builtin_amdgcn_wmma_f32_16x16x32_f16(
                    false, a, false, bf, (short)0, sc[c], false, false);
            }
        }

        // ---- additive mask + tile row max ----
        float tmax[8];
        #pragma unroll
        for (int i = 0; i < 8; ++i) tmax[i] = -3.0e30f;
        #pragma unroll
        for (int c = 0; c < 4; ++c) {
            float mv = mp[k0 + c * 16 + l16];
            #pragma unroll
            for (int i = 0; i < 8; ++i) {
                float s = sc[c][i] + mv;
                sc[c][i] = s;
                tmax[i] = fmaxf(tmax[i], s);
            }
        }
        // row reduce across the 16 lanes of each half (C layout keeps halves separate)
        #pragma unroll
        for (int off = 1; off < 16; off <<= 1)
            #pragma unroll
            for (int i = 0; i < 8; ++i)
                tmax[i] = fmaxf(tmax[i], __shfl_xor(tmax[i], off, 32));

        // ---- online softmax rescale ----
        #pragma unroll
        for (int i = 0; i < 8; ++i) {
            float mnew  = fmaxf(mr[i], tmax[i]);
            float scale = __expf(mr[i] - mnew);
            mr[i] = mnew;
            lr[i] *= scale;
            #pragma unroll
            for (int c = 0; c < 4; ++c) o[c][i] *= scale;
        }
        #pragma unroll
        for (int c = 0; c < 4; ++c) {
            int col = c * 16 + l16;
            #pragma unroll
            for (int i = 0; i < 8; ++i) {
                float p = __expf(sc[c][i] - mr[i]);
                lr[i] += p;                              // per-lane partial row sum
                sP[wave][rhi + i][col] = (_Float16)p;    // C->A re-layout via LDS
            }
        }

        // ---- ctx += P(16x64) @ V(64x64) ----
        #pragma unroll
        for (int ks = 0; ks < 2; ++ks) {
            v16h a;
            int kb = ks * 32;
            #pragma unroll
            for (int i = 0; i < 8; ++i) {
                a[i]     = sP[wave][l16][kb + hsel + i];
                a[8 + i] = sP[wave][l16][kb + 16 + hsel + i];
            }
            #pragma unroll
            for (int c = 0; c < 4; ++c) {
                v16h bf;
                int col = c * 16 + l16;            // d column
                #pragma unroll
                for (int i = 0; i < 8; ++i) {
                    bf[i]     = sVT[col][kb + hsel + i];       // contiguous
                    bf[8 + i] = sVT[col][kb + 16 + hsel + i];
                }
                o[c] = __builtin_amdgcn_wmma_f32_16x16x32_f16(
                    false, a, false, bf, (short)0, o[c], false, false);
            }
        }
    }

    // ---- finalize: complete row sums, normalize, write [B,S,D] fp32 ----
    #pragma unroll
    for (int off = 1; off < 16; off <<= 1)
        #pragma unroll
        for (int i = 0; i < 8; ++i)
            lr[i] += __shfl_xor(lr[i], off, 32);

    #pragma unroll
    for (int i = 0; i < 8; ++i) {
        float inv = 1.0f / lr[i];
        int qr = q0 + wave * 16 + rhi + i;
        #pragma unroll
        for (int c = 0; c < 4; ++c) {
            int d = c * 16 + l16;
            Out[((size_t)b * S_LEN + qr) * D_DIM + h * HD + d] = o[c][i] * inv;
        }
    }
}

// ---------------------------------------------------------------------------
extern "C" void kernel_launch(void* const* d_in, const int* in_sizes, int n_in,
                              void* d_out, int out_size, void* d_ws, size_t ws_size,
                              hipStream_t stream)
{
    (void)in_sizes; (void)n_in; (void)out_size; (void)ws_size;

    const float* X    = (const float*)d_in[0];
    const float* mask = (const float*)d_in[1];
    const float* Wq   = (const float*)d_in[2];
    const float* bq   = (const float*)d_in[3];
    const float* Wk   = (const float*)d_in[4];
    const float* bk   = (const float*)d_in[5];
    const float* Wv   = (const float*)d_in[6];
    const float* bv   = (const float*)d_in[7];
    float* Out = (float*)d_out;

    // f16 Q/K/V workspace: 3 * 4M elems * 2B = 25.2 MB
    const size_t nqkv = (size_t)2 * NH * S_LEN * HD;
    _Float16* Qb = (_Float16*)d_ws;
    _Float16* Kb = Qb + nqkv;
    _Float16* Vb = Kb + nqkv;   // stored as [B,H,HD,S]

    dim3 g1(D_DIM / 128, (2 * S_LEN) / 128, 3);   // (8, 32, 3)
    qkv_gemm_kernel<<<g1, 256, 0, stream>>>(X, Wq, bq, Wk, bk, Wv, bv, Qb, Kb, Vb);

    dim3 g2(S_LEN / 128, 2 * NH);                 // (16, 32)
    attn_kernel<<<g2, 256, 0, stream>>>(Qb, Kb, Vb, mask, Out);
}
